// vit_mem_59906203845058
// MI455X (gfx1250) — matrix operand order repaired and checked by reading, not compiled, not run on hardware
//
#include <hip/hip_runtime.h>
#include <hip/hip_bf16.h>
#include <math.h>

// ---------------------------------------------------------------------------
// top-k retrieval: sim = l2norm(query) @ queue_k^T ; top-32 ; gather k/v rows
// N=512, K=262144, D=256, TOP_N=32, all fp32 in/out.
// f16 WMMA GEMM (16x16x32, f32 accum), double-buffered f16 LDS key staging,
// per-row top-32 via threshold + replace-min, two-kernel partial -> merge.
// ---------------------------------------------------------------------------

typedef __attribute__((ext_vector_type(16))) _Float16 v16h;
typedef __attribute__((ext_vector_type(8)))  _Float16 v8h;
typedef __attribute__((ext_vector_type(8)))  float    v8f;

namespace cfg {
constexpr int N     = 512;
constexpr int K     = 262144;
constexpr int D     = 256;
constexpr int TOPN  = 32;
constexpr int NS    = 64;            // K slices
constexpr int KSL   = K / NS;        // 4096 keys per slice
constexpr int QT    = 128;           // query rows per block (8 waves x 16)
constexpr int NQT   = N / QT;        // 4
constexpr int CHUNK = 16;            // keys per WMMA N-tile step
constexpr int SSTRH = D + 8;         // padded f16 LDS stride (halfs)
constexpr float NEG = -1.0e30f;
}

// ---------------------------------------------------------------------------
// Kernel 1: per (query-tile, K-slice) partial GEMM + top-32
// ---------------------------------------------------------------------------
__global__ __launch_bounds__(256)
void topk_gemm_partial(const float* __restrict__ query,
                       const float* __restrict__ qk,
                       float* __restrict__ wsv,
                       int*   __restrict__ wsi) {
  using namespace cfg;
  __shared__ _Float16 stage[2][CHUNK * SSTRH];   // f16 key chunk, double-buffered
  __shared__ float s_topv[QT * TOPN];
  __shared__ int   s_topi[QT * TOPN];
  __shared__ float s_rowmin[QT];
  __shared__ int   s_rowminpos[QT];

  const int tid  = threadIdx.x;
  const int lane = tid & 31;
  const int wave = tid >> 5;
  const int ks   = blockIdx.x;                 // K slice
  const int qt   = blockIdx.y;                 // query tile
  const int col  = lane & 15;                  // N column within 16x16 tile
  const int hi   = lane >> 4;                  // lane half (A/B K-split)

  // ---- init top-k state -------------------------------------------------
  for (int i = tid; i < QT * TOPN; i += 256) { s_topv[i] = NEG; s_topi[i] = 0; }
  for (int i = tid; i < QT; i += 256) { s_rowmin[i] = NEG; s_rowminpos[i] = 0; }

  // ---- load + l2-normalize A (16 rows x 256 dims f16 in VGPRs) ----------
  // A 16x32 f16 layout: lanes 0-15 hold row=lane, K = {0..7, 16..23} of each
  // 32-block; lanes 16-31 hold row=lane-16, K = {8..15, 24..31}.
  const int qrow = qt * QT + wave * 16 + col;
  const int koff = hi * 8;
  float ss = 0.0f;
  for (int kk = 0; kk < 8; ++kk) {
    const float* p = query + (size_t)qrow * D + kk * 32;
#pragma unroll
    for (int u = 0; u < 8; ++u) { float x = p[koff + u];      ss += x * x; }
#pragma unroll
    for (int u = 0; u < 8; ++u) { float x = p[16 + koff + u]; ss += x * x; }
  }
  ss += __shfl_xor(ss, 16);                    // partner lane holds other 128
  const float rn = 1.0f / fmaxf(sqrtf(ss), 1e-12f);

  v16h a[8];
  for (int kk = 0; kk < 8; ++kk) {
    const float* p = query + (size_t)qrow * D + kk * 32;
#pragma unroll
    for (int u = 0; u < 8; ++u) a[kk][u]     = (_Float16)(p[koff + u] * rn);
#pragma unroll
    for (int u = 0; u < 8; ++u) a[kk][8 + u] = (_Float16)(p[16 + koff + u] * rn);
  }

  // staging helpers: thread t -> key j=t/16, dims (t%16)*16..+15 (16 f32)
  const int sj    = tid >> 4;
  const int sdseg = (tid & 15) * 16;

  auto cvt_store = [&](int buf, const float4 p[4]) {
    v8h h0, h1;
    h0[0]=(_Float16)p[0].x; h0[1]=(_Float16)p[0].y; h0[2]=(_Float16)p[0].z; h0[3]=(_Float16)p[0].w;
    h0[4]=(_Float16)p[1].x; h0[5]=(_Float16)p[1].y; h0[6]=(_Float16)p[1].z; h0[7]=(_Float16)p[1].w;
    h1[0]=(_Float16)p[2].x; h1[1]=(_Float16)p[2].y; h1[2]=(_Float16)p[2].z; h1[3]=(_Float16)p[2].w;
    h1[4]=(_Float16)p[3].x; h1[5]=(_Float16)p[3].y; h1[6]=(_Float16)p[3].z; h1[7]=(_Float16)p[3].w;
    _Float16* dst = &stage[buf][sj * SSTRH + sdseg];
    *reinterpret_cast<v8h*>(dst)     = h0;
    *reinterpret_cast<v8h*>(dst + 8) = h1;
  };

  // ---- prologue: stage chunk 0 ------------------------------------------
  const int nchunk = KSL / CHUNK;
  {
    const int kb0 = ks * KSL;
    const float4* src = reinterpret_cast<const float4*>(qk + (size_t)(kb0 + sj) * D + sdseg);
    float4 p[4] = { src[0], src[1], src[2], src[3] };
    cvt_store(0, p);
  }
  __syncthreads();

  // ---- main loop over 16-key chunks (software-pipelined) -----------------
  for (int c = 0; c < nchunk; ++c) {
    const int kb  = ks * KSL + c * CHUNK;
    const int cur = c & 1;
    const bool havenext = (c + 1 < nchunk);

    // issue global loads for chunk c+1 early (hide behind WMMA)
    float4 pre[4];
    if (havenext) {
      const float4* src =
          reinterpret_cast<const float4*>(qk + (size_t)(kb + CHUNK + sj) * D + sdseg);
      pre[0] = src[0]; pre[1] = src[1]; pre[2] = src[2]; pre[3] = src[3];
      if (c + 2 < nchunk)
        __builtin_prefetch(qk + (size_t)(kb + 2 * CHUNK + sj) * D + sdseg, 0, 0);
    }

    // preload all 8 B fragments, then 8 WMMAs on two accumulator chains
    v16h bfrag[8];
#pragma unroll
    for (int kk = 0; kk < 8; ++kk) {
      // B 32x16 f16 layout: lanes 0-15 -> col, K={0..7,16..23}; lanes 16-31 -> K={8..15,24..31}
      const _Float16* bp = &stage[cur][col * SSTRH + kk * 32 + hi * 8];
      v8h lo = *reinterpret_cast<const v8h*>(bp);
      v8h hh = *reinterpret_cast<const v8h*>(bp + 16);
      bfrag[kk] = __builtin_shufflevector(lo, hh, 0,1,2,3,4,5,6,7,8,9,10,11,12,13,14,15);
    }
    v8f acc0 = {}, acc1 = {};
#pragma unroll
    for (int kk = 0; kk < 8; kk += 2) {
      acc0 = __builtin_amdgcn_wmma_f32_16x16x32_f16(
          false, a[kk],     false, bfrag[kk],     (short)0, acc0, false, false);
      acc1 = __builtin_amdgcn_wmma_f32_16x16x32_f16(
          false, a[kk + 1], false, bfrag[kk + 1], (short)0, acc1, false, false);
    }

    // ---- top-k update (rows disjoint per wave; ballot-serialized insert) --
    const int keyidx = kb + col;
#pragma unroll 1
    for (int r = 0; r < 8; ++r) {
      const float cand = acc0[r] + acc1[r];
      const int   lrow = wave * 16 + r + hi * 8;   // C VGPR r -> rows r / r+8
      const float thr  = s_rowmin[lrow];
      unsigned long long m = __ballot(cand > thr);
      while (m) {
        const int l = __ffsll((unsigned long long)m) - 1;
        m &= m - 1;
        const float v  = __shfl(cand, l);
        const int   rI = __shfl(lrow, l);
        const int   kI = __shfl(keyidx, l);
        if (lane == 0) {
          if (v > s_rowmin[rI]) {                  // recheck vs. live min
            const int p = s_rowminpos[rI];
            s_topv[rI * TOPN + p] = v;
            s_topi[rI * TOPN + p] = kI;
            float mn = s_topv[rI * TOPN]; int mp = 0;
            for (int i = 1; i < TOPN; ++i) {
              const float tv = s_topv[rI * TOPN + i];
              if (tv < mn) { mn = tv; mp = i; }
            }
            s_rowmin[rI] = mn; s_rowminpos[rI] = mp;
          }
        }
      }
    }

    // write chunk c+1 into the other LDS buffer, then one barrier
    if (havenext) cvt_store(cur ^ 1, pre);
    __syncthreads();
  }

  // ---- write partial top-32 lists to workspace ---------------------------
  for (int r = 0; r < 16; ++r) {
    const int lrow = wave * 16 + r;
    const int grow = qt * QT + lrow;
    if (lane < TOPN) {
      const size_t o = ((size_t)grow * NS + ks) * TOPN + lane;
      wsv[o] = s_topv[lrow * TOPN + lane];
      wsi[o] = s_topi[lrow * TOPN + lane];
    }
  }
}

// ---------------------------------------------------------------------------
// Kernel 2: merge NS*32 candidates per row -> final top-32 (descending) and
// gather queue_k / queue_v rows into d_out.
// ---------------------------------------------------------------------------
__global__ __launch_bounds__(256)
void topk_merge_gather(const float* __restrict__ qk,
                       const float* __restrict__ qv,
                       const float* __restrict__ wsv,
                       const int*   __restrict__ wsi,
                       float* __restrict__ out) {
  using namespace cfg;
  constexpr int NC = NS * TOPN;                  // 2048 candidates per row
  __shared__ float cv[NC];
  __shared__ int   ci[NC];
  __shared__ float redv[256];
  __shared__ int   redp[256];
  __shared__ int   sel[TOPN];

  const int row = blockIdx.x;
  const int tid = threadIdx.x;

  for (int i = tid; i < NC; i += 256) {
    cv[i] = wsv[(size_t)row * NC + i];
    ci[i] = wsi[(size_t)row * NC + i];
  }
  __syncthreads();

  for (int s = 0; s < TOPN; ++s) {
    float best = NEG; int bp = -1;
    for (int i = tid; i < NC; i += 256)
      if (cv[i] > best) { best = cv[i]; bp = i; }
    redv[tid] = best; redp[tid] = bp;
    __syncthreads();
    for (int off = 128; off > 0; off >>= 1) {
      if (tid < off && redv[tid + off] > redv[tid]) {
        redv[tid] = redv[tid + off]; redp[tid] = redp[tid + off];
      }
      __syncthreads();
    }
    if (tid == 0) { const int p = redp[0]; sel[s] = ci[p]; cv[p] = NEG; }
    __syncthreads();
  }

  const size_t vhalf = (size_t)N * TOPN * D;     // get_v offset in d_out
  for (int s = 0; s < TOPN; ++s) {
    const size_t src = (size_t)sel[s] * D + tid;
    const size_t dst = ((size_t)row * TOPN + s) * D + tid;
    out[dst]         = qk[src];
    out[vhalf + dst] = qv[src];
  }
}

// ---------------------------------------------------------------------------
extern "C" void kernel_launch(void* const* d_in, const int* in_sizes, int n_in,
                              void* d_out, int out_size, void* d_ws, size_t ws_size,
                              hipStream_t stream) {
  using namespace cfg;
  const float* query = (const float*)d_in[0];
  const float* qk    = (const float*)d_in[1];
  const float* qv    = (const float*)d_in[2];
  float* out = (float*)d_out;

  float* wsv = (float*)d_ws;                         // N*NS*TOPN floats
  int*   wsi = (int*)d_ws + (size_t)N * NS * TOPN;   // N*NS*TOPN ints

  dim3 g1(NS, NQT);
  topk_gemm_partial<<<g1, 256, 0, stream>>>(query, qk, wsv, wsi);
  topk_merge_gather<<<N, 256, 0, stream>>>(qk, qv, wsv, wsi, out);
}
